// MultiHeadAttention_66391604462328
// MI455X (gfx1250) — matrix operand order repaired and checked by reading, not compiled, not run on hardware
//
#include <hip/hip_runtime.h>
#include <hip/hip_bf16.h>

// MHA forward for MI455X (gfx1250, wave32).
// Compute-bound (~206 GFLOP vs ~150MB traffic) => bf16 WMMA everywhere
// (v_wmma_f32_16x16x32_bf16, f32 accumulate). GEMM tiles staged into LDS
// via the Tensor Data Mover (tensor_load_to_lds + s_wait_tensorcnt).

#define BATCH   2
#define SEQ     2048
#define DMODEL  2048
#define NHEAD   16
#define DHEAD   128   // DMODEL / NHEAD
#define MROWS   (BATCH * SEQ)   // 4096

typedef __attribute__((ext_vector_type(16))) __bf16        v16bf;
typedef __attribute__((ext_vector_type(8)))  float         v8f;
typedef __attribute__((ext_vector_type(4)))  unsigned int  u32x4;
typedef __attribute__((ext_vector_type(4)))  int           i32x4;
typedef __attribute__((ext_vector_type(8)))  int           i32x8;

union FragU {
    v16bf v;
    u32x4 q[2];
};

__device__ __forceinline__ unsigned short f32_to_bf16(float f) {
    union { float f; unsigned int u; } c; c.f = f;
    unsigned int u = c.u;
    unsigned int r = (u + 0x7FFFu + ((u >> 16) & 1u)) >> 16;  // RNE
    return (unsigned short)r;
}

// Build a v16bf fragment from two 16-byte LDS chunks.
__device__ __forceinline__ v16bf load_frag(const unsigned short* p0,
                                           const unsigned short* p1) {
    FragU f;
    f.q[0] = *(const u32x4*)p0;
    f.q[1] = *(const u32x4*)p1;
    return f.v;
}

__device__ __forceinline__ float half_reduce_max(float v) {
    #pragma unroll
    for (int off = 1; off < 16; off <<= 1)
        v = fmaxf(v, __shfl_xor(v, off, 32));
    return v;
}
__device__ __forceinline__ float half_reduce_sum(float v) {
    #pragma unroll
    for (int off = 1; off < 16; off <<= 1)
        v += __shfl_xor(v, off, 32);
    return v;
}

// ---------------------------------------------------------------------------
// TDM: DMA a 2D tile (tile_d1 rows x tile_d0 bf16 elems, row stride `stride0`
// elems) from global into LDS at byte offset lds_off (row-major, contiguous).
// D# layout per CDNA5 ISA ch.8 (groups 0/1; groups 2/3 zero for <=2D).
// ---------------------------------------------------------------------------
__device__ __forceinline__ void tdm_load_tile_bf16(
    unsigned lds_off, const unsigned short* gbase, size_t elem_off,
    unsigned tensor_d0, unsigned tensor_d1,
    unsigned tile_d0, unsigned tile_d1, unsigned stride0)
{
    unsigned long long ga =
        (unsigned long long)(uintptr_t)gbase + (unsigned long long)elem_off * 2ull;

    u32x4 g0;
    g0.x = 1u;                                     // count=1, user mode
    g0.y = lds_off;                                // lds_addr [63:32]
    g0.z = (unsigned)(ga & 0xFFFFFFFFu);           // global_addr[31:0]
    g0.w = (unsigned)((ga >> 32) & 0x01FFFFFFu)    // global_addr[56:32]
         | (2u << 30);                             // type=2 ("image")

    i32x8 g1;
    g1[0] = (int)(1u << 16);                       // data_size=1 -> 2 bytes
    g1[1] = (int)((tensor_d0 & 0xFFFFu) << 16);            // tensor_dim0[15:0]
    g1[2] = (int)(((tensor_d0 >> 16) & 0xFFFFu)            // tensor_dim0[31:16]
         |  ((tensor_d1 & 0xFFFFu) << 16));                // tensor_dim1[15:0]
    g1[3] = (int)(((tensor_d1 >> 16) & 0xFFFFu)            // tensor_dim1[31:16]
         |  ((tile_d0 & 0xFFFFu) << 16));                  // tile_dim0
    g1[4] = (int)(tile_d1 & 0xFFFFu);                      // tile_dim1 (tile_dim2=0)
    g1[5] = (int)stride0;                                  // tensor_dim0_stride[31:0]
    g1[6] = 0;                                             // stride0[47:32], dim1_stride lo
    g1[7] = 0;

    i32x4 z4 = {0, 0, 0, 0};
#if defined(__clang_major__) && (__clang_major__ >= 23)
    i32x8 z8 = {0, 0, 0, 0, 0, 0, 0, 0};
    __builtin_amdgcn_tensor_load_to_lds(g0, g1, z4, z4, z8, 0);
#else
    __builtin_amdgcn_tensor_load_to_lds(g0, g1, z4, z4, 0);
#endif
}

// ---------------------------------------------------------------------------
// Elementwise f32 -> bf16 cast
// ---------------------------------------------------------------------------
__global__ void cast_f32_to_bf16_kernel(const float* __restrict__ src,
                                        unsigned short* __restrict__ dst,
                                        int n) {
    int i = blockIdx.x * blockDim.x + threadIdx.x;
    if (i < n) dst[i] = f32_to_bf16(src[i]);
}

// ---------------------------------------------------------------------------
// GEMM: C[M,N] = A[M,K] (bf16) @ B[N,K]^T (bf16) + bias[N] (f32)
// Block: 256 threads = 8 waves; block tile 128x128; wave tile 32x64.
// LDS tiles staged by the Tensor Data Mover (wave 0 issues, all waves use).
// OUT_BF16=1 -> bf16 output, else f32 output.
// ---------------------------------------------------------------------------
template <int OUT_BF16>
__global__ __launch_bounds__(256)
void gemm_bias_wmma(const unsigned short* __restrict__ A,
                    const unsigned short* __restrict__ Bm,
                    const float* __restrict__ bias,
                    void* __restrict__ Cout,
                    int M, int N, int K) {
    __shared__ unsigned short sA[128][32];
    __shared__ unsigned short sB[128][32];

    const int tid  = threadIdx.x;
    const int lane = tid & 31;
    const int l15  = lane & 15;
    const int lh   = lane >> 4;          // lane half (0/1)
    const int w    = tid >> 5;           // wave id 0..7
    const int wm   = w & 3;              // M group (32 rows)
    const int wn   = w >> 2;             // N group (64 cols)

    const int nTiles = N >> 7;
    const int mTile  = blockIdx.x / nTiles;
    const int nTile  = blockIdx.x % nTiles;
    const int mBase  = mTile << 7;
    const int nBase  = nTile << 7;

    const unsigned ldsA = (unsigned)(uintptr_t)&sA[0][0];
    const unsigned ldsB = (unsigned)(uintptr_t)&sB[0][0];

    v8f acc[2][4] = {};

    for (int k0 = 0; k0 < K; k0 += 32) {
        __syncthreads();   // previous iteration done reading sA/sB
        if (tid < 32) {
            // DMA A tile: rows mBase..mBase+127, cols k0..k0+31 of A[M,K].
            tdm_load_tile_bf16(ldsA, A, (size_t)mBase * K + k0,
                               (unsigned)K, (unsigned)M, 32u, 128u, (unsigned)K);
            // DMA B tile: rows nBase..nBase+127, cols k0..k0+31 of B[N,K].
            tdm_load_tile_bf16(ldsB, Bm, (size_t)nBase * K + k0,
                               (unsigned)K, (unsigned)N, 32u, 128u, (unsigned)K);
            __builtin_amdgcn_s_wait_tensorcnt(0);
        }
        __syncthreads();

        // A fragments: row m on lane l15; K runs [lh*8..+7] and [16+lh*8..+7].
        v16bf afrag[2];
        #pragma unroll
        for (int mi = 0; mi < 2; ++mi) {
            const int m = wm * 32 + mi * 16 + l15;
            afrag[mi] = load_frag(&sA[m][lh * 8], &sA[m][16 + lh * 8]);
        }
        // B fragments: column n on lane l15; K runs [lh*8..+7] and [16+lh*8..+7].
        #pragma unroll
        for (int ni = 0; ni < 4; ++ni) {
            const int n = wn * 64 + ni * 16 + l15;
            v16bf bfrag = load_frag(&sB[n][lh * 8], &sB[n][16 + lh * 8]);
            #pragma unroll
            for (int mi = 0; mi < 2; ++mi) {
                acc[mi][ni] = __builtin_amdgcn_wmma_f32_16x16x32_bf16(
                    false, afrag[mi], false, bfrag,
                    (short)0, acc[mi][ni], false, false);
            }
        }
    }

    // Store: f32 C/D layout: VGPR i, lanes 0-15 -> M=i, lanes 16-31 -> M=8+i.
    #pragma unroll
    for (int mi = 0; mi < 2; ++mi) {
        #pragma unroll
        for (int ni = 0; ni < 4; ++ni) {
            #pragma unroll
            for (int i = 0; i < 8; ++i) {
                const int row = mBase + wm * 32 + mi * 16 + lh * 8 + i;
                const int col = nBase + wn * 64 + ni * 16 + l15;
                const float v = acc[mi][ni][i] + bias[col];
                if (OUT_BF16) {
                    ((unsigned short*)Cout)[(size_t)row * N + col] = f32_to_bf16(v);
                } else {
                    ((float*)Cout)[(size_t)row * N + col] = v;
                }
            }
        }
    }
}

// ---------------------------------------------------------------------------
// Flash attention over one (batch, head, 64-row q tile).
// Block: 128 threads = 4 waves; each wave owns 16 q rows.
// Q,K,V,O are bf16 in [B, S, D] layout (head h occupies cols h*128..h*128+127).
// ---------------------------------------------------------------------------
__global__ __launch_bounds__(128)
void flash_attn_wmma(const unsigned short* __restrict__ Qb,
                     const unsigned short* __restrict__ Kb,
                     const unsigned short* __restrict__ Vb,
                     unsigned short* __restrict__ Ob) {
    __shared__ unsigned short sQ[64][128];   // [q][dh]
    __shared__ unsigned short sK[64][128];   // [key][dh]
    __shared__ unsigned short sV[128][64];   // [dh][key] (transposed)
    __shared__ unsigned short sP[64][64];    // [q][key]

    const int tid  = threadIdx.x;
    const int lane = tid & 31;
    const int l15  = lane & 15;
    const int lh   = lane >> 4;
    const int wave = tid >> 5;               // 0..3

    const int qTilesPerHead = SEQ / 64;      // 32
    const int qt = blockIdx.x % qTilesPerHead;
    const int h  = (blockIdx.x / qTilesPerHead) % NHEAD;
    const int b  = blockIdx.x / (qTilesPerHead * NHEAD);
    const int q0 = qt * 64;
    const size_t headCol = (size_t)h * DHEAD;

    // Load Q tile (64 x 128).
    for (int c = tid; c < 1024; c += 128) {
        const int row = c >> 4;
        const int off = (c & 15) * 8;
        *(u32x4*)&sQ[row][off] = *(const u32x4*)
            &Qb[((size_t)(b * SEQ + q0 + row)) * DMODEL + headCol + off];
    }

    const float scale = 0.08838834764831845f;   // 1/sqrt(128)
    float mrow[8], lrow[8];
    v8f   oacc[8] = {};
    #pragma unroll
    for (int i = 0; i < 8; ++i) { mrow[i] = -1e30f; lrow[i] = 0.0f; }

    const int qrowA = wave * 16 + l15;           // A-frag row for this lane

    for (int kt = 0; kt < SEQ / 64; ++kt) {
        __syncthreads();   // previous iter done with sK/sV (also covers sQ load)
        // Load K tile (64x128) and V tile transposed into sV[dh][key].
        for (int c = tid; c < 1024; c += 128) {
            const int key = c >> 4;
            const int off = (c & 15) * 8;
            const size_t g = ((size_t)(b * SEQ + kt * 64 + key)) * DMODEL + headCol + off;
            *(u32x4*)&sK[key][off] = *(const u32x4*)&Kb[g];
            u32x4 tv = *(const u32x4*)&Vb[g];
            const unsigned short* pv = (const unsigned short*)&tv;
            #pragma unroll
            for (int e = 0; e < 8; ++e) sV[off + e][key] = pv[e];
        }
        __syncthreads();

        // S = Q @ K^T for this wave's 16 q rows x 64 keys.
        v8f sacc[4] = {};
        #pragma unroll
        for (int kk = 0; kk < 128; kk += 32) {
            v16bf aq = load_frag(&sQ[qrowA][kk + lh * 8],
                                 &sQ[qrowA][kk + 16 + lh * 8]);
            #pragma unroll
            for (int j = 0; j < 4; ++j) {
                v16bf bk = load_frag(&sK[j * 16 + l15][kk + lh * 8],
                                     &sK[j * 16 + l15][kk + 16 + lh * 8]);
                sacc[j] = __builtin_amdgcn_wmma_f32_16x16x32_bf16(
                    false, aq, false, bk, (short)0, sacc[j], false, false);
            }
        }

        // Online softmax (row r = wave*16 + lh*8 + i lives on this 16-lane half).
        float rmax[8];
        #pragma unroll
        for (int i = 0; i < 8; ++i) rmax[i] = -1e30f;
        #pragma unroll
        for (int j = 0; j < 4; ++j)
            #pragma unroll
            for (int i = 0; i < 8; ++i) {
                sacc[j][i] *= scale;
                rmax[i] = fmaxf(rmax[i], sacc[j][i]);
            }
        #pragma unroll
        for (int i = 0; i < 8; ++i) {
            rmax[i] = half_reduce_max(rmax[i]);
            const float mnew  = fmaxf(mrow[i], rmax[i]);
            const float alpha = __expf(mrow[i] - mnew);
            mrow[i] = mnew;
            lrow[i] *= alpha;
            #pragma unroll
            for (int n = 0; n < 8; ++n) oacc[n][i] *= alpha;
        }
        float rsum[8];
        #pragma unroll
        for (int i = 0; i < 8; ++i) rsum[i] = 0.0f;
        #pragma unroll
        for (int j = 0; j < 4; ++j)
            #pragma unroll
            for (int i = 0; i < 8; ++i) {
                const float p = __expf(sacc[j][i] - mrow[i]);
                sacc[j][i] = p;
                rsum[i] += p;
            }
        #pragma unroll
        for (int i = 0; i < 8; ++i) lrow[i] += half_reduce_sum(rsum[i]);

        // Spill P (bf16) to LDS in A-feedable row-major layout.
        #pragma unroll
        for (int j = 0; j < 4; ++j)
            #pragma unroll
            for (int i = 0; i < 8; ++i)
                sP[wave * 16 + lh * 8 + i][j * 16 + l15] = f32_to_bf16(sacc[j][i]);
        __syncthreads();

        // O += P @ V  (K-dim = 64 keys -> two k-steps of 32).
        #pragma unroll
        for (int kk = 0; kk < 64; kk += 32) {
            v16bf ap = load_frag(&sP[qrowA][kk + lh * 8],
                                 &sP[qrowA][kk + 16 + lh * 8]);
            #pragma unroll
            for (int n = 0; n < 8; ++n) {
                v16bf bv = load_frag(&sV[n * 16 + l15][kk + lh * 8],
                                     &sV[n * 16 + l15][kk + 16 + lh * 8]);
                oacc[n] = __builtin_amdgcn_wmma_f32_16x16x32_bf16(
                    false, ap, false, bv, (short)0, oacc[n], false, false);
            }
        }
    }

    // Normalize and store O (bf16) into [B, S, D] layout.
    float inv[8];
    #pragma unroll
    for (int i = 0; i < 8; ++i) inv[i] = 1.0f / lrow[i];
    #pragma unroll
    for (int n = 0; n < 8; ++n)
        #pragma unroll
        for (int i = 0; i < 8; ++i) {
            const int qrow = q0 + wave * 16 + lh * 8 + i;
            const size_t col = headCol + n * 16 + l15;
            Ob[((size_t)(b * SEQ + qrow)) * DMODEL + col] =
                f32_to_bf16(oacc[n][i] * inv[i]);
        }
}

// ---------------------------------------------------------------------------
extern "C" void kernel_launch(void* const* d_in, const int* in_sizes, int n_in,
                              void* d_out, int out_size, void* d_ws, size_t ws_size,
                              hipStream_t stream) {
    const float* x  = (const float*)d_in[0];
    const float* wq = (const float*)d_in[1];
    const float* bq = (const float*)d_in[2];
    const float* wk = (const float*)d_in[3];
    const float* bk = (const float*)d_in[4];
    const float* wv = (const float*)d_in[5];
    const float* bv = (const float*)d_in[6];
    const float* wo = (const float*)d_in[7];
    const float* bo = (const float*)d_in[8];

    const size_t NX = (size_t)MROWS * DMODEL;   // 8,388,608
    const size_t NW = (size_t)DMODEL * DMODEL;  // 4,194,304

    unsigned short* p   = (unsigned short*)d_ws;
    unsigned short* xb  = p; p += NX;
    unsigned short* wqb = p; p += NW;
    unsigned short* wkb = p; p += NW;
    unsigned short* wvb = p; p += NW;
    unsigned short* wob = p; p += NW;
    unsigned short* Qb  = p; p += NX;
    unsigned short* Kb  = p; p += NX;
    unsigned short* Vb  = p; p += NX;
    unsigned short* Ob  = p; p += NX;

    // 1) Casts to bf16.
    cast_f32_to_bf16_kernel<<<(int)((NX + 255) / 256), 256, 0, stream>>>(x,  xb,  (int)NX);
    cast_f32_to_bf16_kernel<<<(int)((NW + 255) / 256), 256, 0, stream>>>(wq, wqb, (int)NW);
    cast_f32_to_bf16_kernel<<<(int)((NW + 255) / 256), 256, 0, stream>>>(wk, wkb, (int)NW);
    cast_f32_to_bf16_kernel<<<(int)((NW + 255) / 256), 256, 0, stream>>>(wv, wvb, (int)NW);
    cast_f32_to_bf16_kernel<<<(int)((NW + 255) / 256), 256, 0, stream>>>(wo, wob, (int)NW);

    // 2) Projections: Q/K/V = x @ W^T + b  (bf16 out).
    const int gemmBlocks = (MROWS / 128) * (DMODEL / 128);   // 512
    gemm_bias_wmma<1><<<gemmBlocks, 256, 0, stream>>>(xb, wqb, bq, Qb, MROWS, DMODEL, DMODEL);
    gemm_bias_wmma<1><<<gemmBlocks, 256, 0, stream>>>(xb, wkb, bk, Kb, MROWS, DMODEL, DMODEL);
    gemm_bias_wmma<1><<<gemmBlocks, 256, 0, stream>>>(xb, wvb, bv, Vb, MROWS, DMODEL, DMODEL);

    // 3) Attention: B * H * (S/64) blocks of 128 threads.
    const int attnBlocks = BATCH * NHEAD * (SEQ / 64);       // 1024
    flash_attn_wmma<<<attnBlocks, 128, 0, stream>>>(Qb, Kb, Vb, Ob);

    // 4) Output projection: out = attnO @ Wo^T + bo  (f32 out).
    gemm_bias_wmma<0><<<gemmBlocks, 256, 0, stream>>>(Ob, wob, bo, d_out, MROWS, DMODEL, DMODEL);
}
